// RNN_23510650978621
// MI455X (gfx1250) — matrix operand order repaired and checked by reading, not compiled, not run on hardware
//
#include <hip/hip_runtime.h>

// RNN: S=4096 steps of z = relu(h @ Wh.T + bh + pad(x)) * mask, then y = hidden @ Wo.T + bo
#define S_LEN 4096
#define BATCH 64
#define DIN   64
#define HID   256
#define DOUT  64

typedef __attribute__((ext_vector_type(16))) __bf16 v16bf;
typedef __attribute__((ext_vector_type(8)))  __bf16 v8bf;
typedef __attribute__((ext_vector_type(8)))  float  v8f;

// round-to-nearest-even float -> bf16 via bit ops
__device__ __forceinline__ __bf16 f2bf(float f) {
    unsigned u = __builtin_bit_cast(unsigned, f);
    unsigned r = u + 0x7FFFu + ((u >> 16) & 1u);
    unsigned short hs = (unsigned short)(r >> 16);
    return __builtin_bit_cast(__bf16, hs);
}

__device__ __forceinline__ v16bf cat8(v8bf lo, v8bf hi) {
    return __builtin_shufflevector(lo, hi, 0,1,2,3,4,5,6,7,8,9,10,11,12,13,14,15);
}

// ---------------------------------------------------------------------------
// Kernel 1: serial recurrence. One workgroup, 16 waves (512 threads).
// LDS: Wh^T bf16 (128 KB) + double-buffered bf16 h state (64 KB) = 192 KB.
// Per step/wave: 16 A-frag + 64 B-frag ds_load_b128, 32 WMMAs, ONE barrier.
// ---------------------------------------------------------------------------
__global__ void __launch_bounds__(512, 1)
rnn_recurrence(const float* __restrict__ x,
               const float* __restrict__ mask,
               const float* __restrict__ Wh,
               const float* __restrict__ bh,
               float* __restrict__ hidden,   // (S, B, H) fp32
               float* __restrict__ hT)       // (1, B, H) fp32
{
    __shared__ __bf16 sWh[HID * HID];        // 128 KB: sWh[n*HID+k] = Wh[n][k]
    __shared__ __bf16 sh[2][BATCH * HID];    //  64 KB: double-buffered state

    const int tid  = threadIdx.x;
    const int wave = tid >> 5;
    const int lane = tid & 31;

    // stage Wh -> bf16 LDS, zero h0 into buffer 0
    for (int i = tid; i < HID * HID;   i += 512) sWh[i]    = f2bf(Wh[i]);
    for (int i = tid; i < BATCH * HID; i += 512) sh[0][i]  = f2bf(0.0f);

    const int mt = wave & 3;          // row tile: rows 16*mt .. +15
    const int ng = wave >> 2;         // col group: cols 64*ng .. +63 (4 n-tiles)
    const int m0 = mt * 16;
    const int hs = lane >> 4;         // half-select
    const int akoff = hs << 3;        // A: lanes>=16 shift K by 8
    const int bkoff = hs << 3;        // B: lanes>=16 shift K by 8
    const int arow  = m0 + (lane & 15);

    // per-lane epilogue constants
    float biasv[4], maskv[4];
    int   colv[4];
    #pragma unroll
    for (int j = 0; j < 4; ++j) {
        int c = ng * 64 + j * 16 + (lane & 15);
        colv[j]  = c;
        biasv[j] = bh[c];
        maskv[j] = mask[c];
    }
    const bool hasx = (ng == 0);   // only columns 0..63 receive x

    __syncthreads();

    for (int t = 0; t < S_LEN; ++t) {
        const int rb = t & 1;
        const __bf16* hr = sh[rb];
        __bf16*       hw = sh[rb ^ 1];

        // prefetch x[t] injection values; latency hides behind the WMMA chain
        float xv[4][8];
        if (hasx) {
            #pragma unroll
            for (int j = 0; j < 4; ++j) {
                const float* xp = x + ((size_t)t * BATCH) * DIN + colv[j];
                #pragma unroll
                for (int r = 0; r < 8; ++r) {
                    int rr = m0 + r + (hs << 3);
                    xv[j][r] = xp[(size_t)rr * DIN];
                }
            }
        }

        v8f acc[4] = {};
        #pragma unroll
        for (int kc = 0; kc < 8; ++kc) {
            const __bf16* ap = &hr[arow * HID + kc * 32 + akoff];
            v16bf a = cat8(*(const v8bf*)(ap), *(const v8bf*)(ap + 16));
            #pragma unroll
            for (int j = 0; j < 4; ++j) {
                const __bf16* bp = &sWh[colv[j] * HID + kc * 32 + bkoff];
                v16bf b = cat8(*(const v8bf*)(bp), *(const v8bf*)(bp + 16));
                acc[j] = __builtin_amdgcn_wmma_f32_16x16x32_bf16(
                    false, a, false, b, (short)0, acc[j], false, false);
            }
        }

        // epilogue: bias + x + relu + mask; write bf16 state + fp32 trajectory
        float* hout = hidden + (size_t)t * (BATCH * HID);
        #pragma unroll
        for (int j = 0; j < 4; ++j) {
            const int c = colv[j];
            #pragma unroll
            for (int r = 0; r < 8; ++r) {
                const int rr = m0 + r + (hs << 3);
                float v = acc[j][r] + biasv[j];
                if (c < DIN) v += xv[j][r];
                v = fmaxf(v, 0.0f) * maskv[j];
                hw[rr * HID + c]   = f2bf(v);
                hout[rr * HID + c] = v;
                if (t == S_LEN - 1) hT[rr * HID + c] = v;
            }
        }
        __syncthreads();   // h_t complete; buffers swap next step
    }
}

// ---------------------------------------------------------------------------
// Kernel 2: y = hidden @ Wo.T + bo — fully parallel, bandwidth-bound.
// ---------------------------------------------------------------------------
__global__ void __launch_bounds__(256, 2)
rnn_output(const float* __restrict__ hidden,  // (S*B, H) fp32
           const float* __restrict__ Wo,      // (DOUT, H) fp32
           const float* __restrict__ bo,      // (DOUT,)
           float* __restrict__ y)             // (S*B, DOUT) fp32
{
    __shared__ __bf16 sWo[DOUT * HID];  // 32 KB
    __shared__ float  sbo[DOUT];

    const int tid = threadIdx.x;
    for (int i = tid; i < DOUT * HID; i += 256) sWo[i] = f2bf(Wo[i]);
    if (tid < DOUT) sbo[tid] = bo[tid];
    __syncthreads();

    const int wave = tid >> 5;
    const int lane = tid & 31;
    const int hs   = lane >> 4;
    const size_t m0 = (size_t)blockIdx.x * 128 + (size_t)wave * 16;

    v8f acc[4] = {};
    for (int k0 = 0; k0 < HID; k0 += 32) {
        const float* ap = hidden + (m0 + (lane & 15)) * HID + k0 + (hs << 3);
        v16bf a;
        #pragma unroll
        for (int i = 0; i < 8; ++i) a[i]     = f2bf(ap[i]);
        #pragma unroll
        for (int i = 0; i < 8; ++i) a[i + 8] = f2bf(ap[16 + i]);
        #pragma unroll
        for (int j = 0; j < 4; ++j) {
            const __bf16* bp = &sWo[(j * 16 + (lane & 15)) * HID + k0 + (hs << 3)];
            v16bf b = cat8(*(const v8bf*)(bp), *(const v8bf*)(bp + 16));
            acc[j] = __builtin_amdgcn_wmma_f32_16x16x32_bf16(
                false, a, false, b, (short)0, acc[j], false, false);
        }
    }

    #pragma unroll
    for (int j = 0; j < 4; ++j) {
        const int c = j * 16 + (lane & 15);
        #pragma unroll
        for (int r = 0; r < 8; ++r) {
            const size_t rr = m0 + r + (hs << 3);
            y[rr * DOUT + c] = acc[j][r] + sbo[c];
        }
    }
}

// ---------------------------------------------------------------------------
extern "C" void kernel_launch(void* const* d_in, const int* in_sizes, int n_in,
                              void* d_out, int out_size, void* d_ws, size_t ws_size,
                              hipStream_t stream) {
    (void)in_sizes; (void)n_in; (void)out_size; (void)d_ws; (void)ws_size;
    const float* x    = (const float*)d_in[0];
    const float* mask = (const float*)d_in[1];
    const float* Wh   = (const float*)d_in[2];
    const float* bh   = (const float*)d_in[3];
    const float* Wo   = (const float*)d_in[4];
    const float* bo   = (const float*)d_in[5];

    float* hidden = (float*)d_out;                                  // S*B*H
    float* hT     = hidden + (size_t)S_LEN * BATCH * HID;           // B*H
    float* yout   = hT + (size_t)BATCH * HID;                       // S*B*DOUT

    hipLaunchKernelGGL(rnn_recurrence, dim3(1), dim3(512), 0, stream,
                       x, mask, Wh, bh, hidden, hT);
    hipLaunchKernelGGL(rnn_output, dim3((S_LEN * BATCH) / 128), dim3(256), 0, stream,
                       hidden, Wo, bo, yout);
}
